// TemporalGNN_63694365000011
// MI455X (gfx1250) — compile-verified, matrix operand order checked by reading, not run on hardware
//
#include <hip/hip_runtime.h>
#include <hip/hip_bf16.h>
#include <math.h>

#define N_NODES 8192
#define HDIM    256
#define INV_TD  (1.0f / 86400.0f)

typedef __attribute__((ext_vector_type(16))) _Float16 v16h;
typedef __attribute__((ext_vector_type(8)))  _Float16 v8h;
typedef __attribute__((ext_vector_type(8)))  float    v8f;
typedef __attribute__((ext_vector_type(4)))  int      i32x4;

// ---------------------------------------------------------------------------
// CDNA5 async global->LDS copy path (ASYNCcnt tracked), with safe fallback.
// The builtin's pointer params are typed as <4 x i32>* in AS(1)/AS(3).
// ---------------------------------------------------------------------------
#if __has_builtin(__builtin_amdgcn_global_load_async_to_lds_b128) && \
    __has_builtin(__builtin_amdgcn_s_wait_asynccnt)
#define HAVE_ASYNC_LDS 1
#else
#define HAVE_ASYNC_LDS 0
#endif

#if HAVE_ASYNC_LDS
typedef __attribute__((address_space(1))) i32x4 as1_i32x4;
typedef __attribute__((address_space(3))) i32x4 as3_i32x4;
__device__ __forceinline__ void async_cp16(_Float16* ldst, const _Float16* gsrc)
{
  __builtin_amdgcn_global_load_async_to_lds_b128((as1_i32x4*)gsrc, (as3_i32x4*)ldst,
                                                 /*offset=*/0, /*cpol=*/0);
}
#endif

// Copy a contiguous run of `halves` f16 elements into LDS (16B granules).
template <int THREADS>
__device__ __forceinline__ void stage_flat(_Float16* ldst, const _Float16* gsrc,
                                           int halves, int tid)
{
  const int segs = halves >> 3;
  for (int s = tid; s < segs; s += THREADS) {
#if HAVE_ASYNC_LDS
    async_cp16(ldst + s * 8, gsrc + s * 8);
#else
    *(v8h*)(ldst + s * 8) = *(const v8h*)(gsrc + s * 8);
#endif
  }
}

// Copy `rows` rows of 32 f16 (64B) from global (row stride ldg) into LDS [rows][32].
template <int THREADS>
__device__ __forceinline__ void stage_rows32(_Float16* ldst, const _Float16* gsrc,
                                             int rows, int ldg, int tid)
{
  const int segs = rows * 4;                       // 4 x 16B per row
  for (int s = tid; s < segs; s += THREADS) {
    int r = s >> 2, c = (s & 3) * 8;
#if HAVE_ASYNC_LDS
    async_cp16(ldst + r * 32 + c, gsrc + (size_t)r * ldg + c);
#else
    *(v8h*)(ldst + r * 32 + c) = *(const v8h*)(gsrc + (size_t)r * ldg + c);
#endif
  }
}

// ---------------------------------------------------------------------------
// Fragment loader for 16-bit A/B operands of V_WMMA_*_16X16X32 (ISA 7.12.2):
// lane L holds row (L&15); lanes 0-15 hold K=0..7,16..23; lanes 16-31 hold
// K=8..15,24..31 -> two contiguous 16B loads (global_load_b128 / ds_load_b128).
// ---------------------------------------------------------------------------
__device__ __forceinline__ v16h load_frag16(const _Float16* __restrict__ base,
                                            int ld, int lane)
{
  const int m  = lane & 15;
  const int hi = (lane >> 4) & 1;
  const _Float16* p = base + (size_t)m * (size_t)ld + hi * 8;
  v8h lo  = *(const v8h*)(p);
  v8h hi8 = *(const v8h*)(p + 16);
  v16h r;
#pragma unroll
  for (int i = 0; i < 8; ++i) { r[i] = lo[i]; r[i + 8] = hi8[i]; }
  return r;
}

#define WMMA_F32_F16(a, b, c) \
  __builtin_amdgcn_wmma_f32_16x16x32_f16(false, (a), false, (b), (short)0, (c), false, false)

// ---------------------------------------------------------------------------
// Weight convert + transpose: W[k][n] (fp32, 256x256) -> Wt[n][k] (f16)
// ---------------------------------------------------------------------------
__global__ void wtrans_kernel(const float* __restrict__ W, _Float16* __restrict__ Wt)
{
  int i = blockIdx.x * blockDim.x + threadIdx.x;      // 65536 elements
  int k = i >> 8, n = i & 255;
  Wt[(size_t)n * 256 + k] = (_Float16)W[(size_t)k * 256 + n];
}

// ---------------------------------------------------------------------------
// Time encoder fused with residual: h = x + relu((t/tau) * wt + bt), f16 out
// ---------------------------------------------------------------------------
__global__ void prep_time_kernel(const float* __restrict__ x,
                                 const float* __restrict__ ts,
                                 const float* __restrict__ wt,
                                 const float* __restrict__ bt,
                                 _Float16* __restrict__ h)
{
  int i = blockIdx.x * blockDim.x + threadIdx.x;      // N*H
  int n = i >> 8, j = i & 255;
  float tn = ts[n] * INV_TD;
  float te = fmaxf(tn * wt[j] + bt[j], 0.0f);
  h[i] = (_Float16)(x[i] + te);
}

// ---------------------------------------------------------------------------
// C[M,256] = act(A[M,256] @ W + bias), W given transposed (Wt[n][k], f16).
// WG = 256 threads = 8 waves; wave -> 16-row stripe, 16 accum tiles.
// W k-slices (16KB) double-buffered in LDS via async copies, shared by all waves.
// k-loop is FULLY unrolled so af[] stays statically indexed in VGPRs
// (runtime indexing demoted it to scratch in the previous revision).
// ---------------------------------------------------------------------------
__global__ void __launch_bounds__(256)
gemm_f16_wmma(const _Float16* __restrict__ A, const _Float16* __restrict__ Wt,
              const float* __restrict__ bias, _Float16* __restrict__ C,
              int M, int ldc, int relu, int transC)
{
  const int tid  = threadIdx.x;
  const int lane = tid & 31;
  const int wave = tid >> 5;
  const int row0 = blockIdx.x * 128 + wave * 16;

  __shared__ __align__(16) _Float16 lW[2][HDIM * 32];    // 2 x 16KB

  // stage k-slice 0; async copies overlap with the A-fragment preloads below
  stage_rows32<256>(lW[0], Wt, HDIM, HDIM, tid);

  v16h af[8];
#pragma unroll
  for (int s = 0; s < 8; ++s)
    af[s] = load_frag16(A + (size_t)row0 * HDIM + s * 32, HDIM, lane);

  v8f acc[16] = {};

#pragma unroll
  for (int s = 0; s < 8; ++s) {
    const int buf = s & 1;
    if (s + 1 < 8)
      stage_rows32<256>(lW[buf ^ 1], Wt + (s + 1) * 32, HDIM, HDIM, tid);
#if HAVE_ASYNC_LDS
    if (s + 1 < 8) __builtin_amdgcn_s_wait_asynccnt(4);   // 4 in flight for s+1
    else           __builtin_amdgcn_s_wait_asynccnt(0);
#endif
    __syncthreads();

    // 2-deep pipelined tile loop over LDS fragments
    v16h b = load_frag16(lW[buf], 32, lane);
#pragma unroll
    for (int t = 0; t < 16; ++t) {
      v16h bn = b;
      if (t < 15) bn = load_frag16(lW[buf] + (t + 1) * 16 * 32, 32, lane);
      acc[t] = WMMA_F32_F16(af[s], b, acc[t]);
      b = bn;
    }
    __syncthreads();
  }

  const int n_lo = lane & 15;
  const int hi   = (lane >> 4) & 1;
#pragma unroll
  for (int t = 0; t < 16; ++t) {
    int n = t * 16 + n_lo;
    float bn = bias[n];
#pragma unroll
    for (int r = 0; r < 8; ++r) {
      float v = acc[t][r] + bn;
      if (relu) v = fmaxf(v, 0.0f);
      int m = row0 + hi * 8 + r;
      if (!transC) C[(size_t)m * HDIM + n] = (_Float16)v;
      else         C[(size_t)n * ldc + m]  = (_Float16)v;   // V^T
    }
  }
  (void)M;
}

// ---------------------------------------------------------------------------
// Flash attention with fused time-decay bias.
// WG = 128 threads = 4 waves; wave -> 16 query rows. K (32x256) and V^T (256x32)
// chunks double-buffered in LDS via async copies, shared by all 4 waves.
// ---------------------------------------------------------------------------
#define FA_NWAVE 4

__global__ void __launch_bounds__(128)
flash_attn_kernel(const _Float16* __restrict__ Qf, const _Float16* __restrict__ Kf,
                  const _Float16* __restrict__ Vt, const float* __restrict__ ts,
                  float* __restrict__ emb, int N)
{
  const int tid  = threadIdx.x;
  const int lane = tid & 31;
  const int wave = tid >> 5;
  const int q0   = (blockIdx.x * FA_NWAVE + wave) * 16;

  __shared__ __align__(16) _Float16 lK[2][32 * HDIM];     // 2 x 16KB
  __shared__ __align__(16) _Float16 lV[2][HDIM * 32];     // 2 x 16KB
  __shared__ __align__(16) float    sS[FA_NWAVE][16][32];
  __shared__ __align__(16) _Float16 sP[FA_NWAVE][16][32];
  __shared__ float sStat[FA_NWAVE][16][3];                // m_i, l_i, alpha
  __shared__ float sTq[FA_NWAVE][16];

  // stage chunk 0 (async copies overlap with Q-fragment preloads)
  stage_flat<128>(lK[0], Kf, 32 * HDIM, tid);
  stage_rows32<128>(lV[0], Vt, HDIM, N, tid);

  if (lane < 16) {
    sTq[wave][lane]      = ts[q0 + lane];
    sStat[wave][lane][0] = -3.0e38f;
    sStat[wave][lane][1] = 0.0f;
    sStat[wave][lane][2] = 0.0f;
  }

  v16h qf[8];
#pragma unroll
  for (int s = 0; s < 8; ++s)
    qf[s] = load_frag16(Qf + (size_t)q0 * HDIM + s * 32, HDIM, lane);

  v8f acc[16] = {};
  const float inv_scale = 0.0625f;                        // 1/sqrt(256)
  const int n_lo = lane & 15;
  const int hi   = (lane >> 4) & 1;

  const int NC = N / 32;
#pragma unroll 1
  for (int c = 0; c < NC; ++c) {
    const int buf = c & 1;
    const int k0  = c * 32;

    if (c + 1 < NC) {
      stage_flat<128>(lK[buf ^ 1], Kf + (size_t)(k0 + 32) * HDIM, 32 * HDIM, tid);
      stage_rows32<128>(lV[buf ^ 1], Vt + k0 + 32, HDIM, N, tid);
    }
#if HAVE_ASYNC_LDS
    if (c + 1 < NC) __builtin_amdgcn_s_wait_asynccnt(16); // 16 in flight for c+1
    else            __builtin_amdgcn_s_wait_asynccnt(0);
#endif
    __syncthreads();

    const _Float16* Kb = lK[buf];
    const _Float16* Vb = lV[buf];

    // ---- S = Q @ K^T (two 16x16 tiles), 2-deep pipelined LDS reads ---------
    v8f s0 = {}, s1 = {};
    v16h kb0 = load_frag16(Kb, HDIM, lane);
    v16h kb1 = load_frag16(Kb + 16 * HDIM, HDIM, lane);
#pragma unroll
    for (int s = 0; s < 8; ++s) {
      v16h nk0 = kb0, nk1 = kb1;
      if (s < 7) {
        nk0 = load_frag16(Kb + (s + 1) * 32, HDIM, lane);
        nk1 = load_frag16(Kb + 16 * HDIM + (s + 1) * 32, HDIM, lane);
      }
      s0 = WMMA_F32_F16(qf[s], kb0, s0);
      s1 = WMMA_F32_F16(qf[s], kb1, s1);
      kb0 = nk0; kb1 = nk1;
    }

    // ---- scale + exact time-decay bias, spill S to LDS ---------------------
    float tk0 = ts[k0 + n_lo];
    float tk1 = ts[k0 + 16 + n_lo];
#pragma unroll
    for (int r = 0; r < 8; ++r) {
      int m = hi * 8 + r;
      float tq = sTq[wave][m];
      float b0v = logf(expf(-fabsf(tq - tk0) * INV_TD) + 1e-10f);
      float b1v = logf(expf(-fabsf(tq - tk1) * INV_TD) + 1e-10f);
      sS[wave][m][n_lo]      = s0[r] * inv_scale + b0v;
      sS[wave][m][16 + n_lo] = s1[r] * inv_scale + b1v;
    }
    __syncthreads();

    // ---- online softmax bookkeeping (one row per lane 0..15) ---------------
    if (lane < 16) {
      float mOld = sStat[wave][lane][0];
      float rmax = mOld;
#pragma unroll
      for (int j = 0; j < 32; ++j) rmax = fmaxf(rmax, sS[wave][lane][j]);
      float rsum = 0.0f;
#pragma unroll
      for (int j = 0; j < 32; ++j) {
        float p = __expf(sS[wave][lane][j] - rmax);
        sP[wave][lane][j] = (_Float16)p;
        rsum += p;
      }
      float alpha = __expf(mOld - rmax);
      sStat[wave][lane][0] = rmax;
      sStat[wave][lane][1] = sStat[wave][lane][1] * alpha + rsum;
      sStat[wave][lane][2] = alpha;
    }
    __syncthreads();

    // ---- rescale accumulator, then acc += P @ V (pipelined) ----------------
#pragma unroll
    for (int r = 0; r < 8; ++r) {
      float a = sStat[wave][hi * 8 + r][2];
#pragma unroll
      for (int t = 0; t < 16; ++t) acc[t][r] *= a;
    }
    v16h pf = load_frag16(&sP[wave][0][0], 32, lane);
    v16h vb = load_frag16(Vb, 32, lane);
#pragma unroll
    for (int t = 0; t < 16; ++t) {
      v16h vn = vb;
      if (t < 15) vn = load_frag16(Vb + (t + 1) * 16 * 32, 32, lane);
      acc[t] = WMMA_F32_F16(pf, vb, acc[t]);
      vb = vn;
    }
    __syncthreads();
  }

  // ---- normalize and store emb (fp32) --------------------------------------
#pragma unroll
  for (int r = 0; r < 8; ++r) {
    int m = hi * 8 + r;
    float linv = 1.0f / sStat[wave][m][1];
#pragma unroll
    for (int t = 0; t < 16; ++t)
      emb[(size_t)(q0 + m) * HDIM + t * 16 + n_lo] = acc[t][r] * linv;
  }
}

// ---------------------------------------------------------------------------
// Heads: impact = sigmoid(relu(emb@w1+b1)@w2+b2); change = relu(emb@c1+cb1)@c2+cb2
// ---------------------------------------------------------------------------
__global__ void __launch_bounds__(128)
heads_kernel(const float* __restrict__ emb,
             const float* __restrict__ w1, const float* __restrict__ b1,
             const float* __restrict__ w2, const float* __restrict__ b2,
             const float* __restrict__ c1, const float* __restrict__ cb1,
             const float* __restrict__ c2, const float* __restrict__ cb2,
             float* __restrict__ impact, float* __restrict__ change)
{
  const int node = blockIdx.x;
  const int tid  = threadIdx.x;      // 128
  __shared__ float se[256];
  __shared__ float red[128];

  se[tid]       = emb[(size_t)node * HDIM + tid];
  se[tid + 128] = emb[(size_t)node * HDIM + 128 + tid];
  __syncthreads();

  float h1 = b1[tid];
  float h2 = cb1[tid];
  for (int k = 0; k < 256; ++k) {
    float e = se[k];
    h1 += e * w1[(size_t)k * 128 + tid];
    h2 += e * c1[(size_t)k * 128 + tid];
  }
  h1 = fmaxf(h1, 0.0f);
  h2 = fmaxf(h2, 0.0f);

  red[tid] = h1 * w2[tid];
  __syncthreads();
  for (int s = 64; s > 0; s >>= 1) { if (tid < s) red[tid] += red[tid + s]; __syncthreads(); }
  if (tid == 0) impact[node] = 1.0f / (1.0f + __expf(-(red[0] + b2[0])));

  for (int c = 0; c < 3; ++c) {
    __syncthreads();
    red[tid] = h2 * c2[(size_t)tid * 3 + c];
    __syncthreads();
    for (int s = 64; s > 0; s >>= 1) { if (tid < s) red[tid] += red[tid + s]; __syncthreads(); }
    if (tid == 0) change[(size_t)node * 3 + c] = red[0] + cb2[c];
  }
}

// ---------------------------------------------------------------------------
extern "C" void kernel_launch(void* const* d_in, const int* in_sizes, int n_in,
                              void* d_out, int out_size, void* d_ws, size_t ws_size,
                              hipStream_t stream)
{
  (void)in_sizes; (void)n_in; (void)out_size; (void)ws_size;

  const float* x   = (const float*)d_in[0];
  // d_in[1] = edge_index (unused by the reference math)
  const float* ts  = (const float*)d_in[2];
  const float* wt  = (const float*)d_in[3];
  const float* bt  = (const float*)d_in[4];
  const float* Wc  = (const float*)d_in[5];
  const float* bc  = (const float*)d_in[6];
  const float* wq  = (const float*)d_in[7];
  const float* bq  = (const float*)d_in[8];
  const float* wk  = (const float*)d_in[9];
  const float* bk  = (const float*)d_in[10];
  const float* wv  = (const float*)d_in[11];
  const float* bv  = (const float*)d_in[12];
  const float* w1  = (const float*)d_in[13];
  const float* b1  = (const float*)d_in[14];
  const float* w2  = (const float*)d_in[15];
  const float* b2  = (const float*)d_in[16];
  const float* c1  = (const float*)d_in[17];
  const float* cb1 = (const float*)d_in[18];
  const float* c2  = (const float*)d_in[19];
  const float* cb2 = (const float*)d_in[20];

  char* ws = (char*)d_ws;
  const size_t SZ = (size_t)N_NODES * HDIM * sizeof(_Float16);   // 4 MB
  _Float16* hA = (_Float16*)(ws + 0 * SZ);
  _Float16* hB = (_Float16*)(ws + 1 * SZ);
  _Float16* Qf = (_Float16*)(ws + 2 * SZ);
  _Float16* Kf = (_Float16*)(ws + 3 * SZ);
  _Float16* Vt = (_Float16*)(ws + 4 * SZ);                        // [256][8192]
  _Float16* Wt = (_Float16*)(ws + 5 * SZ);                        // 6 x 256x256 f16

  float* emb    = (float*)d_out;
  float* impact = emb + (size_t)N_NODES * HDIM;
  float* change = impact + N_NODES;

  // 1) convert + transpose the six 256x256 weight matrices to f16
  const float* Wsrc[6] = { Wc, Wc + 65536, Wc + 131072, wq, wk, wv };
  for (int i = 0; i < 6; ++i)
    wtrans_kernel<<<256, 256, 0, stream>>>(Wsrc[i], Wt + (size_t)i * 65536);

  // 2) time encoder + residual -> h0 (f16)
  prep_time_kernel<<<(N_NODES * HDIM) / 256, 256, 0, stream>>>(x, ts, wt, bt, hA);

  // 3) conv stack (relu)
  gemm_f16_wmma<<<N_NODES / 128, 256, 0, stream>>>(hA, Wt + 0 * 65536, bc + 0,   hB, N_NODES, 0, 1, 0);
  gemm_f16_wmma<<<N_NODES / 128, 256, 0, stream>>>(hB, Wt + 1 * 65536, bc + 256, hA, N_NODES, 0, 1, 0);
  gemm_f16_wmma<<<N_NODES / 128, 256, 0, stream>>>(hA, Wt + 2 * 65536, bc + 512, hB, N_NODES, 0, 1, 0);

  // 4) Q, K row-major; V stored transposed for the PV matmul
  gemm_f16_wmma<<<N_NODES / 128, 256, 0, stream>>>(hB, Wt + 3 * 65536, bq, Qf, N_NODES, 0,       0, 0);
  gemm_f16_wmma<<<N_NODES / 128, 256, 0, stream>>>(hB, Wt + 4 * 65536, bk, Kf, N_NODES, 0,       0, 0);
  gemm_f16_wmma<<<N_NODES / 128, 256, 0, stream>>>(hB, Wt + 5 * 65536, bv, Vt, N_NODES, N_NODES, 0, 1);

  // 5) streaming attention with fused time-decay bias -> emb (fp32, d_out)
  flash_attn_kernel<<<N_NODES / 64, 128, 0, stream>>>(Qf, Kf, Vt, ts, emb, N_NODES);

  // 6) prediction heads
  heads_kernel<<<N_NODES, 128, 0, stream>>>(emb, w1, b1, w2, b2, c1, cb1, c2, cb2,
                                            impact, change);
}
